// Encoder_57853209477193
// MI455X (gfx1250) — matrix operand order repaired and checked by reading, not compiled, not run on hardware
//
#include <hip/hip_runtime.h>

// ---------------- problem constants ----------------
constexpr int NN  = 32768;
constexpr int SS  = 16;
constexpr int EE  = 512;
constexpr int HH  = 8;
constexpr int DD  = 64;
constexpr int HID = 768;
constexpr int E2  = 1024;   // 2*E
constexpr int MAILC = EE + 2; // 514 columns per mail row

// ---------------- WMMA / TDM types ----------------
typedef __attribute__((ext_vector_type(16))) __bf16 v16bf;
typedef __attribute__((ext_vector_type(8)))  float  v8f;
typedef __attribute__((ext_vector_type(4)))  unsigned int v4u;
typedef __attribute__((ext_vector_type(8)))  int v8i;
typedef __attribute__((ext_vector_type(4)))  int v4i;

union FragBF { uint4 u[2]; v16bf v; };

static __device__ __forceinline__ unsigned short f2bf(float f) {
  unsigned u = __float_as_uint(f);
  u += 0x7FFFu + ((u >> 16) & 1u);
  return (unsigned short)(u >> 16);
}
static __device__ __forceinline__ float bflo(unsigned u) { return __uint_as_float(u << 16); }
static __device__ __forceinline__ float bfhi(unsigned u) { return __uint_as_float(u & 0xffff0000u); }

// A fragment: 16x32 bf16. base points at (row0, k0). lane&15 = row, lane>>4 picks K halves.
static __device__ __forceinline__ v16bf frag_a(const unsigned short* base, int stride, int lane) {
  int row = lane & 15, hi = lane >> 4;
  const unsigned short* p = base + (size_t)row * stride + hi * 8;
  FragBF f;
  f.u[0] = *(const uint4*)(p);        // K = k0 + hi*8 + 0..7
  f.u[1] = *(const uint4*)(p + 16);   // K = k0 + 16 + hi*8 + 0..7
  return f.v;
}

// B fragment: 32x16 bf16 where B[k][j] = W[col0+j][k0+k] with W row-major [out][in].
static __device__ __forceinline__ v16bf frag_b(const unsigned short* wbase, int stride, int lane) {
  int j = lane & 15, hi = lane >> 4;
  const unsigned short* p = wbase + (size_t)j * stride + hi * 8;
  FragBF f;
  f.u[0] = *(const uint4*)(p);
  f.u[1] = *(const uint4*)(p + 16);
  return f.v;
}

static __device__ __forceinline__ v8f wmma_bf16(v16bf a, v16bf b, v8f c) {
  return __builtin_amdgcn_wmma_f32_16x16x32_bf16(false, a, false, b, (short)0, c, false, false);
}

// ---------------- Tensor Data Mover: 2D tile load (global -> LDS) ----------------
// Builds D# per CDNA5 ISA ch.8: group0 = {count=1, lds_addr, global_addr(57b), type=2},
// group1 = {mask=0, data_size, tensor_dim0/1, tile_dim0/1, tensor_dim0_stride}.
// data_size_log: 0=1B,1=2B,2=4B,3=8B. Dims/strides in data_size units.
static __device__ __forceinline__ void tdm_load_2d(unsigned lds_off, const void* gaddr,
                                                   unsigned tensor_d0, unsigned tensor_d1,
                                                   unsigned tile_d0, unsigned tile_d1,
                                                   unsigned long long d0_stride,
                                                   unsigned data_size_log) {
  unsigned long long ga = (unsigned long long)(size_t)gaddr;
  v4u g0 = {0u, 0u, 0u, 0u};
  g0[0] = 1u;                                           // count=1 (valid user descriptor)
  g0[1] = lds_off;                                      // lds_addr [63:32]
  g0[2] = (unsigned)ga;                                 // global_addr[31:0]
  g0[3] = (unsigned)((ga >> 32) & 0x01FFFFFFu) | (2u << 30); // addr[56:32] | type=2
  v8i g1 = {0, 0, 0, 0, 0, 0, 0, 0};
  g1[0] = (int)(data_size_log << 16);                   // workgroup_mask=0, data_size
  g1[1] = (int)((tensor_d0 & 0xFFFFu) << 16);           // tensor_dim0[15:0] at bits[63:48]
  g1[2] = (int)((tensor_d0 >> 16) | ((tensor_d1 & 0xFFFFu) << 16)); // dim0 hi | dim1 lo
  g1[3] = (int)((tensor_d1 >> 16) | (tile_d0 << 16));   // dim1 hi | tile_dim0
  g1[4] = (int)(tile_d1);                               // tile_dim1 | tile_dim2=0
  g1[5] = (int)(unsigned)d0_stride;                     // tensor_dim0_stride[31:0]
  g1[6] = (int)(unsigned)((d0_stride >> 32) & 0xFFFFu); // stride[47:32] | dim1_stride lo=0
  g1[7] = 0;
  v4i gz = {0, 0, 0, 0};
#if defined(__clang_major__) && (__clang_major__ >= 23)
  v8i gz8 = {0, 0, 0, 0, 0, 0, 0, 0};
  __builtin_amdgcn_tensor_load_to_lds(g0, g1, gz, gz, gz8, 0);
#else
  __builtin_amdgcn_tensor_load_to_lds(g0, g1, gz, gz, 0);
#endif
}

static __device__ __forceinline__ unsigned lds_off_of(const void* p) {
  // generic pointers to LDS carry the LDS byte offset in the low 32 bits
  return (unsigned)(size_t)p;
}

// ---------------- kernel: f32 -> bf16 weight conversion (optional transpose) ----------------
__global__ void convert_weights_kernel(const float* __restrict__ src, unsigned short* __restrict__ dst,
                                       int rows, int cols, int transpose) {
  int idx = blockIdx.x * blockDim.x + threadIdx.x;
  if (idx >= rows * cols) return;
  int r = idx / cols, c = idx - r * cols;
  unsigned short b = f2bf(src[idx]);
  if (transpose) dst[(size_t)c * rows + r] = b;
  else           dst[idx] = b;
}

// ---------------- kernel: feat_t = feat + cos(dt*w + b) ----------------
__global__ void feat_time_kernel(const float* __restrict__ feat, const float* __restrict__ ts,
                                 const float* __restrict__ lu, const float* __restrict__ time_w,
                                 const float* __restrict__ time_b, float* __restrict__ featt_f32,
                                 unsigned short* __restrict__ featt_bf) {
  int idx = blockIdx.x * blockDim.x + threadIdx.x;
  if (idx >= NN * EE) return;
  int n = idx >> 9, e = idx & (EE - 1);
  float dt = ts[n] - lu[n];
  float v = feat[idx] + __cosf(dt * time_w[e] + time_b[e]);
  featt_f32[idx] = v;
  featt_bf[idx]  = f2bf(v);
}

// ---------------- kernel: generic bf16 GEMM  out = A @ W^T + bias ----------------
// A: [M][K] bf16 row-major, W: [Ncols][K] bf16 row-major.
// A tile (16 x K) staged once per workgroup via the Tensor Data Mover; B streams from L2.
// mode 0: bf16 out, 1: f32 out, 2: bf16 out + relu
__global__ __launch_bounds__(256) void gemm_bf16_kernel(const unsigned short* __restrict__ A,
                                                        const unsigned short* __restrict__ W,
                                                        const float* __restrict__ bias,
                                                        void* __restrict__ out,
                                                        int M, int Ncols, int K, int mode) {
  __shared__ unsigned short shA[16 * 1024];  // max K = 1024
  int lane = threadIdx.x & 31, wave = threadIdx.x >> 5;
  int rowt = blockIdx.x;
  int colbase = blockIdx.y * 128 + wave * 16;

  if (wave == 0) {  // TDM issue is wave-level (EXEC ignored)
    tdm_load_2d(lds_off_of(shA), A + (size_t)rowt * 16 * K,
                (unsigned)K, (unsigned)M, (unsigned)K, 16u,
                (unsigned long long)K, 1u /*2B*/);
    __builtin_amdgcn_s_wait_tensorcnt(0);
  }
  __syncthreads();

  if (colbase < Ncols) {
    const unsigned short* Wcol = W + (size_t)colbase * K;
    v8f acc = {};
    for (int k0 = 0; k0 < K; k0 += 32) {
      __builtin_prefetch(Wcol + k0 + 256, 0, 0);   // next B chunk -> global_prefetch
      v16bf a = frag_a(shA + k0, K, lane);         // ds_load_b128 from staged tile
      v16bf b = frag_b(Wcol + k0, K, lane);
      acc = wmma_bf16(a, b, acc);
    }
    int col = colbase + (lane & 15);
    int hi = lane >> 4;
    float bcol = bias ? bias[col] : 0.0f;
    for (int r = 0; r < 8; ++r) {
      int row = rowt * 16 + hi * 8 + r;
      float v = acc[r] + bcol;
      if (mode == 2) v = fmaxf(v, 0.0f);
      if (mode == 1) ((float*)out)[(size_t)row * Ncols + col] = v;
      else ((unsigned short*)out)[(size_t)row * Ncols + col] = f2bf(v);
    }
  }
}

// ---------------- fused attention kernel (8 nodes / workgroup) ----------------
constexpr int OFF_Q    = 8 * 16 * 512 * 2;            // joined: [8][16][512] bf16 -> 131072
constexpr int OFF_R    = OFF_Q + 16 * 512 * 2;        // q: [16][512] bf16 (rows 8..15 slack)
constexpr int OFF_SC   = OFF_R + 16 * 8 * 512 * 2;    // r/m: [16][8][512] bf16 (nodes 0..7 valid)
constexpr int OFF_TIME = OFF_SC + 8 * 8 * 16 * 4;     // scores: [8][8][16] f32
constexpr int OFF_ORD  = OFF_TIME + 8 * 16 * 4;       // time: [8][16] f32
constexpr int OFF_TS   = OFF_ORD + 8 * 16 * 4;        // order: [8][16] i32
constexpr int SH_ATTN  = OFF_TS + 8 * 4;              // total 283680 B (<320KB/WG on CDNA5)

__global__ __launch_bounds__(256) void attn_kernel(const float* __restrict__ mail,
                                                   const float* __restrict__ ts,
                                                   const float* __restrict__ time_w,
                                                   const float* __restrict__ time_b,
                                                   const float* __restrict__ pos_emb,
                                                   const unsigned short* __restrict__ qb,
                                                   const unsigned short* __restrict__ WkT,
                                                   const unsigned short* __restrict__ Wv,
                                                   const float* __restrict__ bv,
                                                   unsigned short* __restrict__ ctx_out,
                                                   float* __restrict__ attn_w_out) {
  extern __shared__ char smem[];
  unsigned short* sh_joined = (unsigned short*)(smem);
  unsigned short* sh_q      = (unsigned short*)(smem + OFF_Q);
  unsigned short* sh_r      = (unsigned short*)(smem + OFF_R);   // later reused as m
  float*          sh_sc     = (float*)(smem + OFF_SC);
  float*          sh_time   = (float*)(smem + OFF_TIME);
  int*            sh_ord    = (int*)(smem + OFF_ORD);
  float*          sh_ts     = (float*)(smem + OFF_TS);

  const int tid = threadIdx.x;
  const int lane = tid & 31, wave = tid >> 5;
  const int n0 = blockIdx.x * 8;

  // stage q rows for these 8 nodes via TDM (8 x 512 bf16 tile)
  if (wave == 0) {
    tdm_load_2d(lds_off_of(sh_q), qb + (size_t)n0 * EE,
                (unsigned)EE, (unsigned)NN, (unsigned)EE, 8u,
                (unsigned long long)EE, 1u /*2B*/);
    __builtin_amdgcn_s_wait_tensorcnt(0);
  }
  // phase 0: mail times + node ts
  if (tid < 128) {
    int i = tid >> 4, s = tid & 15;
    sh_time[tid] = mail[(size_t)(n0 + i) * (SS * MAILC) + s * MAILC + EE];
    if (s == 0) sh_ts[i] = ts[n0 + i];
  }
  __syncthreads();

  // phase 0b: stable argsort ranks -> order[rank]=slot
  if (tid < 128) {
    int i = tid >> 4, s = tid & 15;
    float myt = sh_time[i * 16 + s];
    int rank = 0;
    for (int t = 0; t < 16; ++t) {
      float ot = sh_time[i * 16 + t];
      rank += (ot < myt) || (ot == myt && t < s);
    }
    sh_ord[i * 16 + rank] = s;
  }
  __syncthreads();

  // phase 1: build joined = mail + cos(dt*w+b) + pos_emb[order[s]]
  for (int idx = tid; idx < 8 * SS * EE; idx += 256) {
    int i = idx >> 13;
    int s = (idx >> 9) & 15;
    int e = idx & (EE - 1);
    float dt = sh_ts[i] - sh_time[i * 16 + s];
    float v = mail[(size_t)(n0 + i) * (SS * MAILC) + s * MAILC + e]
            + __cosf(dt * time_w[e] + time_b[e])
            + pos_emb[sh_ord[i * 16 + s] * EE + e];
    sh_joined[idx] = f2bf(v);
  }
  __syncthreads();

  // phase 2 (WMMA): r[i][h][:] = sum_d q[i][h*64+d] * Wk[h*64+d][:]
  for (int t = wave; t < 256; t += 8) {
    int h = t >> 5, jt = t & 31;
    v8f acc = {};
    for (int k0 = 0; k0 < DD; k0 += 32) {
      v16bf a = frag_a(sh_q + h * DD + k0, EE, lane);
      v16bf b = frag_b(WkT + (size_t)(jt * 16) * EE + h * DD + k0, EE, lane);
      acc = wmma_bf16(a, b, acc);
    }
    int hi = lane >> 4, col = jt * 16 + (lane & 15);
    for (int r = 0; r < 8; ++r) {
      int i = hi * 8 + r;  // rows 8..15 are slack, region allocated
      sh_r[((size_t)i * 8 + h) * EE + col] = f2bf(acc[r]);
    }
  }
  __syncthreads();

  // phase 3: scores[i][h][s] = joined[i][s][:] . r[i][h][:] / sqrt(d)
  for (int t = tid; t < 8 * HH * SS; t += 256) {
    int i = t >> 7, h = (t >> 4) & 7, s = t & 15;
    const unsigned short* jp = sh_joined + ((size_t)i * 16 + s) * EE;
    const unsigned short* rp = sh_r + ((size_t)i * 8 + h) * EE;
    float acc = 0.f;
    for (int e = 0; e < EE; e += 8) {
      uint4 ju = *(const uint4*)(jp + e);
      uint4 ru = *(const uint4*)(rp + e);
      acc += bflo(ju.x) * bflo(ru.x) + bfhi(ju.x) * bfhi(ru.x);
      acc += bflo(ju.y) * bflo(ru.y) + bfhi(ju.y) * bfhi(ru.y);
      acc += bflo(ju.z) * bflo(ru.z) + bfhi(ju.z) * bfhi(ru.z);
      acc += bflo(ju.w) * bflo(ru.w) + bfhi(ju.w) * bfhi(ru.w);
    }
    sh_sc[t] = acc * 0.125f;  // 1/sqrt(64)
  }
  __syncthreads();

  // softmax over s per (i,h)
  if (tid < 64) {
    int i = tid >> 3, h = tid & 7;
    float* sc = sh_sc + (i * 8 + h) * 16;
    float mx = sc[0];
    for (int s = 1; s < 16; ++s) mx = fmaxf(mx, sc[s]);
    float ex[16], sum = 0.f;
    for (int s = 0; s < 16; ++s) { ex[s] = __expf(sc[s] - mx); sum += ex[s]; }
    float inv = 1.0f / sum;
    for (int s = 0; s < 16; ++s) sc[s] = ex[s] * inv;
  }
  __syncthreads();

  // attn_weight = mean over heads (straight to output region 2)
  if (tid < 128) {
    int i = tid >> 4, s = tid & 15;
    float a = 0.f;
    for (int h = 0; h < 8; ++h) a += sh_sc[(i * 8 + h) * 16 + s];
    attn_w_out[(size_t)(n0 + i) * SS + s] = a * 0.125f;
  }
  __syncthreads();

  // phase 4: m[i][h][:] = sum_s attn * joined[i][s][:]  (overwrites sh_r)
  for (int t = tid; t < 8 * HH * (EE / 4); t += 256) {
    int i = t >> 10, h = (t >> 7) & 7, e = (t & 127) * 4;
    const float* at = sh_sc + (i * 8 + h) * 16;
    float a0 = 0, a1 = 0, a2 = 0, a3 = 0;
    for (int s = 0; s < 16; ++s) {
      uint2 ju = *(const uint2*)(sh_joined + ((size_t)i * 16 + s) * EE + e);
      float w = at[s];
      a0 += w * bflo(ju.x); a1 += w * bfhi(ju.x);
      a2 += w * bflo(ju.y); a3 += w * bfhi(ju.y);
    }
    unsigned short* mp = sh_r + ((size_t)i * 8 + h) * EE + e;
    mp[0] = f2bf(a0); mp[1] = f2bf(a1); mp[2] = f2bf(a2); mp[3] = f2bf(a3);
  }
  __syncthreads();

  // phase 5 (WMMA): ctx[i][h*64+oc] = m[i][h][:] . Wv[h*64+oc][:] + bv
  for (int t = wave; t < 32; t += 8) {
    int h = t >> 2, ct = t & 3;
    v8f acc = {};
    for (int k0 = 0; k0 < EE; k0 += 32) {
      v16bf a = frag_a(sh_r + h * EE + k0, HH * EE, lane);  // row stride = 8*512
      v16bf b = frag_b(Wv + (size_t)(h * DD + ct * 16) * EE + k0, EE, lane);
      acc = wmma_bf16(a, b, acc);
    }
    int hi = lane >> 4;
    int oc = h * DD + ct * 16 + (lane & 15);
    if (hi == 0) {
      float bvv = bv[oc];
      for (int r = 0; r < 8; ++r)
        ctx_out[(size_t)(n0 + r) * EE + oc] = f2bf(acc[r] + bvv);
    }
  }
}

// ---------------- fused Wo GEMM + concat + LayerNorm ----------------
constexpr int WO_SHA  = 16 * E2 * 4;            // sx: [16][1024] f32
constexpr int SH_WOLN = WO_SHA + 16 * EE * 2;   // + staged ctx tile [16][512] bf16

__global__ __launch_bounds__(256) void wo_ln_kernel(const unsigned short* __restrict__ ctxb,
                                                    const unsigned short* __restrict__ Wo,
                                                    const float* __restrict__ bo,
                                                    const float* __restrict__ featt,
                                                    const float* __restrict__ ln_g,
                                                    const float* __restrict__ ln_b,
                                                    unsigned short* __restrict__ xn_out) {
  extern __shared__ char smem[];
  float* sx = (float*)smem;                                  // [16][1024]
  unsigned short* shA = (unsigned short*)(smem + WO_SHA);    // [16][512] bf16
  const int tid = threadIdx.x;
  const int lane = tid & 31, wave = tid >> 5;
  const int rowt = blockIdx.x;

  // stage ctx tile (16 x 512 bf16) via TDM
  if (wave == 0) {
    tdm_load_2d(lds_off_of(shA), ctxb + (size_t)rowt * 16 * EE,
                (unsigned)EE, (unsigned)NN, (unsigned)EE, 16u,
                (unsigned long long)EE, 1u /*2B*/);
    __builtin_amdgcn_s_wait_tensorcnt(0);
  }
  // x[:, 0:512] = feat_t
  for (int idx = tid; idx < 16 * EE; idx += 256) {
    int r = idx >> 9, e = idx & (EE - 1);
    sx[r * E2 + e] = featt[(size_t)(rowt * 16 + r) * EE + e];
  }
  __syncthreads();

  // x[:, 512:1024] = ctx @ Wo^T + bo  (each wave: 4 col-tiles of 16)
  v8f acc[4] = {};
  for (int k0 = 0; k0 < EE; k0 += 32) {
    v16bf a = frag_a(shA + k0, EE, lane);
    for (int ct = 0; ct < 4; ++ct) {
      v16bf b = frag_b(Wo + (size_t)(wave * 64 + ct * 16) * EE + k0, EE, lane);
      acc[ct] = wmma_bf16(a, b, acc[ct]);
    }
  }
  int hi = lane >> 4;
  for (int ct = 0; ct < 4; ++ct) {
    int col = wave * 64 + ct * 16 + (lane & 15);
    float bcol = bo[col];
    for (int r = 0; r < 8; ++r)
      sx[(hi * 8 + r) * E2 + EE + col] = acc[ct][r] + bcol;
  }
  __syncthreads();

  // LayerNorm over 1024, 2 rows per wave (wave32 shuffle reduction)
  for (int rr = 0; rr < 2; ++rr) {
    int r = wave * 2 + rr;
    float s = 0.f, s2 = 0.f;
    for (int j = lane; j < E2; j += 32) { float v = sx[r * E2 + j]; s += v; s2 += v * v; }
    for (int off = 16; off; off >>= 1) {
      s  += __shfl_xor(s, off, 32);
      s2 += __shfl_xor(s2, off, 32);
    }
    float mu  = s * (1.0f / E2);
    float var = s2 * (1.0f / E2) - mu * mu;
    float inv = rsqrtf(var + 1e-5f);
    size_t grow = (size_t)(rowt * 16 + r);
    for (int j = lane; j < E2; j += 32) {
      float v = (sx[r * E2 + j] - mu) * inv * ln_g[j] + ln_b[j];
      xn_out[grow * E2 + j] = f2bf(v);
    }
  }
}

// ---------------- host ----------------
extern "C" void kernel_launch(void* const* d_in, const int* in_sizes, int n_in,
                              void* d_out, int out_size, void* d_ws, size_t ws_size,
                              hipStream_t stream) {
  const float* feat    = (const float*)d_in[0];
  const float* mail    = (const float*)d_in[1];
  const float* ts      = (const float*)d_in[2];
  const float* lu      = (const float*)d_in[3];
  const float* Wq      = (const float*)d_in[4];
  const float* bq      = (const float*)d_in[5];
  const float* Wk      = (const float*)d_in[6];
  // d_in[7] = bk: constant per (n,h) across s -> drops out of softmax, unused
  const float* Wv      = (const float*)d_in[8];
  const float* bv      = (const float*)d_in[9];
  const float* Wo      = (const float*)d_in[10];
  const float* bo      = (const float*)d_in[11];
  const float* time_w  = (const float*)d_in[12];
  const float* time_b  = (const float*)d_in[13];
  const float* pos_emb = (const float*)d_in[14];
  const float* ln_g    = (const float*)d_in[15];
  const float* ln_b    = (const float*)d_in[16];
  const float* W1      = (const float*)d_in[17];
  const float* b1      = (const float*)d_in[18];
  const float* W2      = (const float*)d_in[19];
  const float* b2      = (const float*)d_in[20];

  char* ws = (char*)d_ws;
  size_t o = 0;
  auto alloc = [&](size_t bytes) { size_t r = o; o += (bytes + 255) & ~(size_t)255; return r; };
  size_t o_wq  = alloc((size_t)EE * EE * 2);
  size_t o_wkT = alloc((size_t)EE * EE * 2);   // Wk stored transposed: [in][out]
  size_t o_wv  = alloc((size_t)EE * EE * 2);
  size_t o_wo  = alloc((size_t)EE * EE * 2);
  size_t o_w1  = alloc((size_t)HID * E2 * 2);
  size_t o_w2  = alloc((size_t)EE * HID * 2);
  size_t o_ftf = alloc((size_t)NN * EE * 4);   // feat_t f32
  size_t o_ftb = alloc((size_t)NN * EE * 2);   // feat_t bf16
  size_t o_qb  = alloc((size_t)NN * EE * 2);   // q bf16
  size_t o_ctx = alloc((size_t)NN * EE * 2);   // ctx bf16
  size_t o_xn  = alloc((size_t)NN * E2 * 2);   // layernormed x bf16
  size_t o_hid = alloc((size_t)NN * HID * 2);  // relu hidden bf16

  unsigned short* wq_b  = (unsigned short*)(ws + o_wq);
  unsigned short* wkT_b = (unsigned short*)(ws + o_wkT);
  unsigned short* wv_b  = (unsigned short*)(ws + o_wv);
  unsigned short* wo_b  = (unsigned short*)(ws + o_wo);
  unsigned short* w1_b  = (unsigned short*)(ws + o_w1);
  unsigned short* w2_b  = (unsigned short*)(ws + o_w2);
  float*          ftf   = (float*)(ws + o_ftf);
  unsigned short* ftb   = (unsigned short*)(ws + o_ftb);
  unsigned short* qb    = (unsigned short*)(ws + o_qb);
  unsigned short* ctxb  = (unsigned short*)(ws + o_ctx);
  unsigned short* xnb   = (unsigned short*)(ws + o_xn);
  unsigned short* hidb  = (unsigned short*)(ws + o_hid);

  auto cvt = [&](const float* src, unsigned short* dst, int r, int c, int tr) {
    int tot = r * c;
    convert_weights_kernel<<<(tot + 255) / 256, 256, 0, stream>>>(src, dst, r, c, tr);
  };
  cvt(Wq, wq_b, EE, EE, 0);
  cvt(Wk, wkT_b, EE, EE, 1);   // transpose -> B-fragment loads contiguous
  cvt(Wv, wv_b, EE, EE, 0);
  cvt(Wo, wo_b, EE, EE, 0);
  cvt(W1, w1_b, HID, E2, 0);
  cvt(W2, w2_b, EE, HID, 0);

  feat_time_kernel<<<(NN * EE) / 256, 256, 0, stream>>>(feat, ts, lu, time_w, time_b, ftf, ftb);

  // q = feat_t @ Wq^T + bq  (bf16 out)
  gemm_bf16_kernel<<<dim3(NN / 16, EE / 128), 256, 0, stream>>>(ftb, wq_b, bq, qb, NN, EE, EE, 0);

  // fused attention: ctx (bf16) + attn_weight (f32, second output region)
  float* attn_w_out = (float*)d_out + (size_t)NN * EE;
  attn_kernel<<<NN / 8, 256, SH_ATTN, stream>>>(mail, ts, time_w, time_b, pos_emb,
                                                qb, wkT_b, wv_b, bv, ctxb, attn_w_out);

  // attn_output = ctx @ Wo^T + bo ; x = [feat_t | attn_output]; LayerNorm -> xn bf16
  wo_ln_kernel<<<NN / 16, 256, SH_WOLN, stream>>>(ctxb, wo_b, bo, ftf, ln_g, ln_b, xnb);

  // hid = relu(xn @ W1^T + b1)
  gemm_bf16_kernel<<<dim3(NN / 16, HID / 128), 256, 0, stream>>>(xnb, w1_b, b1, hidb, NN, HID, E2, 2);

  // out = hid @ W2^T + b2  (f32, first output region)
  gemm_bf16_kernel<<<dim3(NN / 16, EE / 128), 256, 0, stream>>>(hidb, w2_b, b2, d_out, NN, EE, HID, 1);
}
